// MultiHeadAttentionAutoencoder_8211977470482
// MI455X (gfx1250) — matrix operand order repaired and checked by reading, not compiled, not run on hardware
//
#include <hip/hip_runtime.h>
#include <hip/hip_bf16.h>

typedef __attribute__((ext_vector_type(16))) __bf16 v16bf;
typedef __attribute__((ext_vector_type(8)))  float  v8f;

union Frag16 {
  v16bf v;
  uint4 q[2];
};

static __device__ __forceinline__ unsigned short f2bf(float f) {
  unsigned int u = __float_as_uint(f);
  unsigned int r = u + 0x7fffu + ((u >> 16) & 1u);   // round-to-nearest-even
  return (unsigned short)(r >> 16);
}

// 16-byte global -> LDS direct DMA (no VGPR staging), tracked by ASYNCcnt.
static __device__ __forceinline__ void async_copy16(void* lds, const void* gptr) {
  unsigned l = (unsigned)(uintptr_t)lds;   // generic LDS addr: low 32 bits are the LDS offset
  asm volatile("global_load_async_to_lds_b128 %0, %1, off"
               :: "v"(l), "v"(gptr) : "memory");
}

// ---------------- fp32 -> bf16 conversion ----------------
__global__ void cvt_f32_bf16(const float* __restrict__ in,
                             unsigned short* __restrict__ out, size_t n) {
  size_t i = (size_t)blockIdx.x * blockDim.x + threadIdx.x;
  size_t stride = (size_t)gridDim.x * blockDim.x;
  for (; i < n; i += stride) out[i] = f2bf(in[i]);
}

// ---------------- generic bf16 WMMA GEMM:  C = A @ W^T ----------------
// A: [M x K] bf16 row-major, W: [N x K] bf16 row-major (torch Linear layout)
// Double-buffered A tile in LDS filled by async global->LDS DMA.
#define BM 128
#define BN 128
#define KSTEP 32

template <bool NGUARD>
__global__ __launch_bounds__(256) void gemm_bf16(
    const unsigned short* __restrict__ A, const unsigned short* __restrict__ W,
    float* __restrict__ Cf, unsigned short* __restrict__ Cb,
    int M, int N, int K, int relu)
{
  __shared__ __align__(16) unsigned short sA[2][BM][KSTEP];   // 2 x 8 KB
  const int tid  = threadIdx.x;
  const int lane = tid & 31;
  const int wave = tid >> 5;          // 8 waves
  const int wrow = wave >> 2;         // 0..1  (64 rows each)
  const int wcol = wave & 3;          // 0..3  (32 cols each)
  const int ln   = lane & 15;
  const int half = lane >> 4;
  const int bm = blockIdx.y * BM;
  const int bn = blockIdx.x * BN;
  const int wn = bn + wcol * 32;

  v8f acc[4][2] = {};

  const int ar = tid >> 1;            // 0..127
  const int ac = (tid & 1) * 16;      // 0 or 16
  const unsigned short* aptr = A + (size_t)(bm + ar) * K + ac;

  // prologue: async-stage tile 0 into buffer 0
  async_copy16(&sA[0][ar][ac],     aptr);
  async_copy16(&sA[0][ar][ac + 8], aptr + 8);

  for (int k0 = 0; k0 < K; k0 += KSTEP) {
    const int buf = (k0 / KSTEP) & 1;
    // kick off next tile's DMA into the other buffer (overlaps this tile's WMMAs)
    if (k0 + KSTEP < K) {
      async_copy16(&sA[buf ^ 1][ar][ac],     aptr + k0 + KSTEP);
      async_copy16(&sA[buf ^ 1][ar][ac + 8], aptr + k0 + KSTEP + 8);
      // async loads complete in order: <=2 outstanding means tile k0 has landed
      asm volatile("s_wait_asynccnt 0x2" ::: "memory");
    } else {
      asm volatile("s_wait_asynccnt 0x0" ::: "memory");
    }
    __syncthreads();                  // whole tile visible to all waves

    // A fragments for this wave's 4 M-subtiles
    Frag16 af[4];
#pragma unroll
    for (int mt = 0; mt < 4; ++mt) {
      const unsigned short* p = &sA[buf][wrow*64 + mt*16 + ln][half*8];
      af[mt].q[0] = *(const uint4*)(p);        // K {0..7} / {8..15}
      af[mt].q[1] = *(const uint4*)(p + 16);   // K {16..23} / {24..31}
    }

#pragma unroll
    for (int nt = 0; nt < 2; ++nt) {
      Frag16 bfragv;
      const int col = wn + nt*16 + ln;
      if (!NGUARD || col < N) {
        const unsigned short* wp = W + (size_t)col * K + k0 + half*8;
        bfragv.q[0] = *(const uint4*)(wp);       // K 0..7 / 8..15
        bfragv.q[1] = *(const uint4*)(wp + 16);  // K 16..23 / 24..31
      } else {
        bfragv.q[0] = make_uint4(0, 0, 0, 0);
        bfragv.q[1] = make_uint4(0, 0, 0, 0);
      }
#pragma unroll
      for (int mt = 0; mt < 4; ++mt) {
        acc[mt][nt] = __builtin_amdgcn_wmma_f32_16x16x32_bf16(
            false, af[mt].v, false, bfragv.v, (short)0, acc[mt][nt], false, false);
      }
    }
    __syncthreads();                  // done reading: safe for next DMA overwrite
  }

  // epilogue
#pragma unroll
  for (int mt = 0; mt < 4; ++mt) {
#pragma unroll
    for (int nt = 0; nt < 2; ++nt) {
#pragma unroll
      for (int r = 0; r < 8; ++r) {
        const int row = bm + wrow*64 + mt*16 + half*8 + r;
        const int col = wn + nt*16 + ln;
        if (!NGUARD || col < N) {
          float vv = acc[mt][nt][r];
          if (relu) vv = vv > 0.f ? vv : 0.f;
          const size_t idx = (size_t)row * N + col;
          if (Cf) Cf[idx] = vv;
          if (Cb) Cb[idx] = f2bf(vv);
        }
      }
    }
  }
}

// ---------------- flash-style multi-head attention ----------------
// qkv: [B*S][3*E] bf16 (q | k | v packed per token row)
// ctx: [B*S][E]   bf16
#define SEQ 1024
#define DH  128

__global__ __launch_bounds__(128) void attention_kernel(
    const unsigned short* __restrict__ qkv, unsigned short* __restrict__ ctx)
{
  __shared__ __align__(16) unsigned short sVt[DH][64];    // transposed V tile [dh][krow], 16 KB
  __shared__ __align__(16) unsigned short sP[4][16][64];  // per-wave P tile, 8 KB
  const int qt   = blockIdx.x;    // 0..15  (64-row Q tile)
  const int bh   = blockIdx.y;    // 0..127
  const int b    = bh >> 3;
  const int h    = bh & 7;
  const int tid  = threadIdx.x;
  const int lane = tid & 31;
  const int wave = tid >> 5;      // 4 waves, 16 Q rows each
  const int ln   = lane & 15;
  const int half = lane >> 4;
  const size_t RS = 3 * SEQ;      // qkv row stride
  const unsigned short* base = qkv + (size_t)b * SEQ * RS;

  // Preload Q fragments (A operand): rows qt*64 + wave*16 + ln, Dh=128 -> 4 K-steps
  Frag16 qf[4];
  {
    const unsigned short* qrow = base + (size_t)(qt*64 + wave*16 + ln) * RS + h * DH;
#pragma unroll
    for (int ks = 0; ks < 4; ++ks) {
      const unsigned short* p = qrow + ks*32 + half*8;
      qf[ks].q[0] = *(const uint4*)(p);
      qf[ks].q[1] = *(const uint4*)(p + 16);
    }
  }

  float m_run[8], l_run[8];
#pragma unroll
  for (int r = 0; r < 8; ++r) { m_run[r] = -3.0e38f; l_run[r] = 0.f; }
  v8f ov[8] = {};
  const float scale = 0.088388347648318447f;   // 1/sqrt(128)

  for (int kt = 0; kt < 16; ++kt) {
    __syncthreads();   // WAR on sVt from previous iteration
    // cooperative transposed stage of the 64x128 V tile
    {
      const int vr = tid >> 1;            // 0..63
      const int c0 = (tid & 1) * 64;      // dh half
      const unsigned short* vp = base + (size_t)(kt*64 + vr) * RS + 2*SEQ + h*DH + c0;
#pragma unroll
      for (int i = 0; i < 64; i += 8) {
        uint4 u = *(const uint4*)(vp + i);
        unsigned short e[8];
        *(uint4*)e = u;
#pragma unroll
        for (int j = 0; j < 8; ++j) sVt[c0 + i + j][vr] = e[j];
      }
    }
    __syncthreads();   // RAW on sVt

    // S = Q K^T for this wave's 16 rows x 64 cols (K frags straight from global)
    v8f sacc[4] = {};
#pragma unroll
    for (int nt = 0; nt < 4; ++nt) {
#pragma unroll
      for (int ks = 0; ks < 4; ++ks) {
        Frag16 kb;
        const unsigned short* kp =
            base + (size_t)(kt*64 + nt*16 + ln) * RS + SEQ + h*DH + ks*32 + half*8;
        kb.q[0] = *(const uint4*)(kp);
        kb.q[1] = *(const uint4*)(kp + 16);
        sacc[nt] = __builtin_amdgcn_wmma_f32_16x16x32_bf16(
            false, qf[ks].v, false, kb.v, (short)0, sacc[nt], false, false);
      }
    }

    // streaming softmax: rows r+8*half live across the 16 lanes of each half-wave
    float mcur[8];
#pragma unroll
    for (int r = 0; r < 8; ++r) {
      float v = fmaxf(fmaxf(sacc[0][r], sacc[1][r]), fmaxf(sacc[2][r], sacc[3][r])) * scale;
#pragma unroll
      for (int off = 1; off < 16; off <<= 1) v = fmaxf(v, __shfl_xor(v, off, 32));
      mcur[r] = v;
    }
    float alpha[8], lsum[8];
#pragma unroll
    for (int r = 0; r < 8; ++r) {
      float mnew = fmaxf(m_run[r], mcur[r]);
      alpha[r] = __expf(m_run[r] - mnew);
      m_run[r] = mnew;
      lsum[r] = 0.f;
    }
#pragma unroll
    for (int nt = 0; nt < 4; ++nt) {
#pragma unroll
      for (int r = 0; r < 8; ++r) {
        float p = __expf(sacc[nt][r] * scale - m_run[r]);
        lsum[r] += p;
        sP[wave][half*8 + r][nt*16 + ln] = f2bf(p);
      }
    }
#pragma unroll
    for (int r = 0; r < 8; ++r) {
      float s = lsum[r];
#pragma unroll
      for (int off = 1; off < 16; off <<= 1) s += __shfl_xor(s, off, 32);
      l_run[r] = l_run[r] * alpha[r] + s;
    }
#pragma unroll
    for (int nt = 0; nt < 8; ++nt)
#pragma unroll
      for (int r = 0; r < 8; ++r) ov[nt][r] *= alpha[r];

    asm volatile("s_wait_dscnt 0" ::: "memory");   // same-wave sP store->load ordering

    // O += P V   (2 K-steps of 32 over the 64 K-rows)
#pragma unroll
    for (int ks2 = 0; ks2 < 2; ++ks2) {
      Frag16 pa;
      const unsigned short* pp = &sP[wave][ln][ks2*32 + half*8];
      pa.q[0] = *(const uint4*)(pp);
      pa.q[1] = *(const uint4*)(pp + 16);
#pragma unroll
      for (int nt = 0; nt < 8; ++nt) {
        Frag16 vb;
        const unsigned short* vp = &sVt[nt*16 + ln][ks2*32 + half*8];
        vb.q[0] = *(const uint4*)(vp);
        vb.q[1] = *(const uint4*)(vp + 16);
        ov[nt] = __builtin_amdgcn_wmma_f32_16x16x32_bf16(
            false, pa.v, false, vb.v, (short)0, ov[nt], false, false);
      }
    }
  }

  // finalize and store ctx (bf16)
#pragma unroll
  for (int nt = 0; nt < 8; ++nt) {
#pragma unroll
    for (int r = 0; r < 8; ++r) {
      const int row = qt*64 + wave*16 + half*8 + r;
      const int col = h*DH + nt*16 + ln;
      const float v = ov[nt][r] / l_run[r];
      ctx[((size_t)b * SEQ + row) * SEQ + col] = f2bf(v);
    }
  }
}

// ---------------- residual + LayerNorm (no affine) ----------------
__global__ __launch_bounds__(256) void layernorm_residual(
    const float* __restrict__ attn, const float* __restrict__ x,
    unsigned short* __restrict__ hout)
{
  __shared__ float rs1[8], rs2[8];
  __shared__ float s_mu, s_rstd;
  const int row = blockIdx.x;
  const float* a  = attn + (size_t)row * 1024;
  const float* xr = x    + (size_t)row * 1024;
  float vals[4];
  float s = 0.f, s2 = 0.f;
#pragma unroll
  for (int i = 0; i < 4; ++i) {
    float v = a[threadIdx.x + i*256] + xr[threadIdx.x + i*256];
    vals[i] = v; s += v; s2 += v * v;
  }
#pragma unroll
  for (int off = 16; off >= 1; off >>= 1) {
    s  += __shfl_xor(s,  off, 32);
    s2 += __shfl_xor(s2, off, 32);
  }
  if ((threadIdx.x & 31) == 0) { rs1[threadIdx.x >> 5] = s; rs2[threadIdx.x >> 5] = s2; }
  __syncthreads();
  if (threadIdx.x == 0) {
    float S = 0.f, S2 = 0.f;
    for (int i = 0; i < 8; ++i) { S += rs1[i]; S2 += rs2[i]; }
    float mu  = S * (1.f / 1024.f);
    float var = S2 * (1.f / 1024.f) - mu * mu;
    s_mu = mu; s_rstd = rsqrtf(var + 1e-5f);
  }
  __syncthreads();
#pragma unroll
  for (int i = 0; i < 4; ++i)
    hout[(size_t)row*1024 + threadIdx.x + i*256] = f2bf((vals[i] - s_mu) * s_rstd);
}

// ---------------- launcher ----------------
extern "C" void kernel_launch(void* const* d_in, const int* in_sizes, int n_in,
                              void* d_out, int out_size, void* d_ws, size_t ws_size,
                              hipStream_t stream)
{
  (void)in_sizes; (void)n_in; (void)out_size; (void)ws_size;
  const float* x     = (const float*)d_in[0];
  const float* w_in  = (const float*)d_in[1];
  const float* w_out = (const float*)d_in[2];
  const float* w_e1  = (const float*)d_in[3];
  const float* w_e2  = (const float*)d_in[4];
  const float* w_d1  = (const float*)d_in[5];
  const float* w_d2  = (const float*)d_in[6];

  float* recon = (float*)d_out;                  // [16384 x 1024]
  float* enc_f = recon + (size_t)16384 * 1024;   // [16384 x 64]

  char* ws = (char*)d_ws;
  size_t off = 0;
  auto takebf = [&](size_t elems) -> unsigned short* {
    unsigned short* p = (unsigned short*)(ws + off);
    off += ((elems * 2 + 255) / 256) * 256;
    return p;
  };
  unsigned short* xb    = takebf((size_t)16384 * 1024);
  unsigned short* winb  = takebf((size_t)3072 * 1024);
  unsigned short* woutb = takebf((size_t)1024 * 1024);
  unsigned short* we1b  = takebf((size_t)128 * 1024);
  unsigned short* we2b  = takebf((size_t)64 * 128);
  unsigned short* wd1b  = takebf((size_t)128 * 64);
  unsigned short* wd2b  = takebf((size_t)1024 * 128);
  unsigned short* qkv   = takebf((size_t)16384 * 3072);
  unsigned short* ctx   = takebf((size_t)16384 * 1024);
  unsigned short* hbuf  = takebf((size_t)16384 * 1024);
  unsigned short* e1b   = takebf((size_t)16384 * 128);
  unsigned short* encb  = takebf((size_t)16384 * 64);
  unsigned short* d1b   = takebf((size_t)16384 * 128);
  float* attn_out = (float*)(ws + off);
  off += (size_t)16384 * 1024 * 4;

  // fp32 -> bf16
  cvt_f32_bf16<<<dim3(2048), dim3(256), 0, stream>>>(x,     xb,    (size_t)16384 * 1024);
  cvt_f32_bf16<<<dim3(512),  dim3(256), 0, stream>>>(w_in,  winb,  (size_t)3072 * 1024);
  cvt_f32_bf16<<<dim3(256),  dim3(256), 0, stream>>>(w_out, woutb, (size_t)1024 * 1024);
  cvt_f32_bf16<<<dim3(64),   dim3(256), 0, stream>>>(w_e1,  we1b,  (size_t)128 * 1024);
  cvt_f32_bf16<<<dim3(8),    dim3(256), 0, stream>>>(w_e2,  we2b,  (size_t)64 * 128);
  cvt_f32_bf16<<<dim3(8),    dim3(256), 0, stream>>>(w_d1,  wd1b,  (size_t)128 * 64);
  cvt_f32_bf16<<<dim3(64),   dim3(256), 0, stream>>>(w_d2,  wd2b,  (size_t)1024 * 128);

  // qkv = x @ w_in^T          (M=16384, N=3072, K=1024)
  gemm_bf16<false><<<dim3(3072/128, 16384/128), dim3(256), 0, stream>>>(
      xb, winb, nullptr, qkv, 16384, 3072, 1024, 0);
  // multi-head attention -> ctx
  attention_kernel<<<dim3(16, 128), dim3(128), 0, stream>>>(qkv, ctx);
  // attn_out = ctx @ w_out^T  (fp32 for LayerNorm)
  gemm_bf16<false><<<dim3(1024/128, 16384/128), dim3(256), 0, stream>>>(
      ctx, woutb, attn_out, nullptr, 16384, 1024, 1024, 0);
  // h = LN(attn_out + x)
  layernorm_residual<<<dim3(16384), dim3(256), 0, stream>>>(attn_out, x, hbuf);
  // e1 = relu(h @ w_e1^T)     (N=128, K=1024)
  gemm_bf16<false><<<dim3(1, 16384/128), dim3(256), 0, stream>>>(
      hbuf, we1b, nullptr, e1b, 16384, 128, 1024, 1);
  // encoded = relu(e1 @ w_e2^T)  -> fp32 (output #2) + bf16   (N=64 -> guarded)
  gemm_bf16<true><<<dim3(1, 16384/128), dim3(256), 0, stream>>>(
      e1b, we2b, enc_f, encb, 16384, 64, 128, 1);
  // d1 = relu(encoded @ w_d1^T)
  gemm_bf16<false><<<dim3(1, 16384/128), dim3(256), 0, stream>>>(
      encb, wd1b, nullptr, d1b, 16384, 128, 64, 1);
  // reconstruction = d1 @ w_d2^T  -> fp32 (output #1)
  gemm_bf16<false><<<dim3(1024/128, 16384/128), dim3(256), 0, stream>>>(
      d1b, wd2b, recon, nullptr, 16384, 1024, 128, 0);
}
